// LSTM_skip_43052752175248
// MI455X (gfx1250) — compile-verified, matrix operand order checked by reading, not run on hardware
//
#include <hip/hip_runtime.h>

// ---------------------------------------------------------------------------
// LSTM with windowed skip connection, CDNA5 (gfx1250, wave32, WMMA bf16).
//
// Phase 1: swz_init_kernel -- bf16 + WMMA-fragment-swizzle of all weights into
//            ws; fold gate biases; seed the h-ring (entry 31 = h0); zero
//            barrier counters.  Re-run every launch (deterministic).
// Phase 2: xg_kernel       -- xg = x @ W_ih^T + (b_ih+b_hh); bf16 WMMA
//            16x16x32; result stored in bf16 *C-fragment layout* so the
//            recurrent kernel reads it with one b128 per lane.
// Phase 3: rec_kernel      -- persistent: 16 batch-groups x 8 h-slice WGs.
//            Weights LDS-resident (full W_skip^T 128KB + W_hh^T slice 64KB,
//            fragment-swizzled, loaded via global_load_async_to_lds_b128).
//            Hidden state lives in a global bf16 fragment-layout ring
//            (32 entries): next-step h and the t-24 skip operand both come
//            from the ring; the skip operand is async-prefetched into
//            double-buffered LDS one step ahead (ASYNCcnt).  One global
//            atomic barrier per timestep per 8-WG group; c in registers.
// ---------------------------------------------------------------------------

#define Bb 256
#define Tt 1024
#define Cc 128
#define Hh 256
#define G4 1024
#define WIN 24

// workspace layout (bytes)
#define OFF_WIH   0u          // 256 frags * 1KB = 262144
#define OFF_WHH   262144u     // 512 frags * 1KB = 524288
#define OFF_WSK   786432u     // 128 frags * 1KB = 131072
#define OFF_BIAS  917504u     // 1024 f32        = 4096
#define OFF_CTR   921600u     // 16 u32
#define OFF_RING  1048576u    // 32 ent * 16 grp * 8KB = 4MB
#define OFF_XG    5242880u    // T*16*64*256 bf16 = 512MB (C-fragment layout)

typedef __attribute__((ext_vector_type(16))) __bf16 v16bf;
typedef __attribute__((ext_vector_type(8)))  float  v8f;

union V16U { v16bf v; unsigned short u[16]; uint4 q[2]; };
union V8U  { uint4 q; unsigned short u[8]; };

__device__ __forceinline__ unsigned short f2bf(float f) {
  unsigned u = __builtin_bit_cast(unsigned, f);
  u += 0x7FFFu + ((u >> 16) & 1u);        // round-to-nearest-even
  return (unsigned short)(u >> 16);
}
__device__ __forceinline__ float bf2f(unsigned short s) {
  unsigned u = ((unsigned)s) << 16;
  return __builtin_bit_cast(float, u);
}
__device__ __forceinline__ float bfw_lo(unsigned w) {   // low bf16 of a dword
  return __builtin_bit_cast(float, w << 16);
}
__device__ __forceinline__ float bfw_hi(unsigned w) {   // high bf16 of a dword
  return __builtin_bit_cast(float, w & 0xffff0000u);
}
// K offset within a 32-wide K-tile for (lane, value-slot) of a 16-bit A/B
// WMMA fragment (ISA 7.12.2): lanes 16-31 -> +8, slots 8-15 -> +16.
__device__ __forceinline__ int kk_of(int lane, int v) {
  return ((lane >> 4) << 3) + ((v >> 3) << 4) + (v & 7);
}

__device__ __forceinline__ v8f wmma_bf16(v16bf a, v16bf b, v8f c) {
  return __builtin_amdgcn_wmma_f32_16x16x32_bf16(false, a, false, b,
                                                 (short)0, c, false, false);
}

__device__ __forceinline__ float fsigmoid(float x) {
  return 1.0f / (1.0f + __expf(-x));
}
__device__ __forceinline__ float ftanh(float x) {
  float e = __expf(2.0f * x);
  return (e - 1.0f) / (e + 1.0f);
}

// CDNA5 async global->LDS copy, 16B per lane, tracked by ASYNCcnt.
__device__ __forceinline__ void async_copy_b128(void* lds_dst, const void* gsrc) {
  unsigned ldsoff = (unsigned)(size_t)lds_dst;           // low 32b = LDS offset
  unsigned long long ga = (unsigned long long)(size_t)gsrc;
  asm volatile("global_load_async_to_lds_b128 %0, %1, off"
               :: "v"(ldsoff), "v"(ga) : "memory");
}
__device__ __forceinline__ void wait_async0() {
  asm volatile("s_wait_asynccnt 0" ::: "memory");
}

// ---------------------------------------------------------------------------
// Phase 1: weight swizzle + per-launch state init (525328 work items).
// ---------------------------------------------------------------------------
__global__ __launch_bounds__(256) void swz_init_kernel(
    const float* __restrict__ W_ih, const float* __restrict__ W_hh,
    const float* __restrict__ b_ih, const float* __restrict__ b_hh,
    const float* __restrict__ W_skip, const float* __restrict__ h0,
    unsigned char* __restrict__ ws) {
  unsigned short* swih = (unsigned short*)(ws + OFF_WIH);
  unsigned short* swhh = (unsigned short*)(ws + OFF_WHH);
  unsigned short* swsk = (unsigned short*)(ws + OFF_WSK);
  float*          bias = (float*)(ws + OFF_BIAS);
  unsigned short* ring = (unsigned short*)(ws + OFF_RING);
  unsigned int*   ctr  = (unsigned int*)(ws + OFF_CTR);

  unsigned int idx = blockIdx.x * 256u + threadIdx.x;
  if (idx < 131072u) {                       // W_ih^T  [C x 4H], 64 nt x 4 kt
    int frag = idx >> 9, pos = idx & 511;
    int lane = pos >> 4, v = pos & 15;
    int nt = frag >> 2, kt = frag & 3;
    int k = kt * 32 + kk_of(lane, v);
    int n = nt * 16 + (lane & 15);
    swih[idx] = f2bf(W_ih[n * Cc + k]);
  } else if (idx < 393216u) {                // W_hh^T  [H x 4H], 64 nt x 8 kt
    unsigned int i2 = idx - 131072u;
    int frag = i2 >> 9, pos = i2 & 511;
    int lane = pos >> 4, v = pos & 15;
    int nt = frag >> 3, kt = frag & 7;
    int k = kt * 32 + kk_of(lane, v);
    int n = nt * 16 + (lane & 15);
    swhh[i2] = f2bf(W_hh[n * Hh + k]);
  } else if (idx < 458752u) {                // W_skip^T [H x H], 16 nt x 8 kt
    unsigned int i3 = idx - 393216u;
    int frag = i3 >> 9, pos = i3 & 511;
    int lane = pos >> 4, v = pos & 15;
    int nt = frag >> 3, kt = frag & 7;
    int k = kt * 32 + kk_of(lane, v);
    int n = nt * 16 + (lane & 15);
    swsk[i3] = f2bf(W_skip[n * Hh + k]);
  } else if (idx < 459776u) {                // folded gate bias
    int g = (int)(idx - 458752u);
    bias[g] = b_ih[g] + b_hh[g];
  } else if (idx < 525312u) {                // ring entry 31 = h(-1) = h0
    int i5 = (int)(idx - 459776u);           // 16 groups * 4096 shorts
    int g2 = i5 >> 12, rem = i5 & 4095;
    int kt = rem >> 9, pos = rem & 511;
    int lane = pos >> 4, v = pos & 15;
    int k = kt * 32 + kk_of(lane, v);
    ring[((long)(31 * 16 + g2)) * 4096 + rem] = f2bf(h0[k]);
  } else if (idx < 525328u) {                // barrier counters
    ctr[idx - 525312u] = 0u;
  }
}

// ---------------------------------------------------------------------------
// Phase 2: xg = x @ W_ih^T + bias, bf16 C-fragment-layout output.
// grid = T * 16 btiles * 4 colgroups, 256 threads (8 waves, 2 ntiles each).
// ---------------------------------------------------------------------------
__global__ __launch_bounds__(256) void xg_kernel(
    const float* __restrict__ x, unsigned char* __restrict__ ws) {
  extern __shared__ unsigned short sA[];     // 4 A-frags * 512 shorts = 4KB
  const unsigned short* swih = (const unsigned short*)(ws + OFF_WIH);
  const float*          bias = (const float*)(ws + OFF_BIAS);
  unsigned short*       xgf  = (unsigned short*)(ws + OFF_XG);

  int bid = blockIdx.x;
  int cg  = bid & 3;            // 256-col gate group
  int bt  = (bid >> 2) & 15;    // batch tile
  int t   = bid >> 6;
  int tid = threadIdx.x;
  int b0  = bt * 16;

  // stage A = x[b0:b0+16, t, 0:128] as 4 bf16 fragments
#pragma unroll
  for (int i = 0; i < 8; ++i) {
    int idx = tid * 8 + i;                   // 0..2047
    int m = idx >> 7, k = idx & 127;
    float xv = x[((long)(b0 + m) * Tt + t) * Cc + k];
    int lane = (m & 15) + (((k >> 3) & 1) << 4);
    int v    = (((k >> 4) & 1) << 3) + (k & 7);
    sA[(k >> 5) * 512 + lane * 16 + v] = f2bf(xv);
  }
  __syncthreads();

  int w = tid >> 5, lane = tid & 31;
#pragma unroll
  for (int s = 0; s < 2; ++s) {
    int NT = cg * 16 + 2 * w + s;            // global ntile 0..63
    v8f acc = {};
#pragma unroll
    for (int kt = 0; kt < 4; ++kt) {
      v16bf a  = *(const v16bf*)(sA + kt * 512 + lane * 16);
      v16bf bm = *(const v16bf*)(swih + (unsigned)(NT * 4 + kt) * 512u + lane * 16);
      acc = wmma_bf16(a, bm, acc);
    }
    float bv = bias[NT * 16 + (lane & 15)];
    V8U uv;
#pragma unroll
    for (int r = 0; r < 8; ++r) uv.u[r] = f2bf(acc[r] + bv);
    // C-fragment layout: one 16B store per lane
    long ent = (((long)t * 16 + bt) * 64 + NT) * 256 + lane * 8;
    *(uint4*)(xgf + ent) = uv.q;
  }
}

// ---------------------------------------------------------------------------
// Phase 3: persistent recurrent kernel.
// grid = 128 WGs: group g = blockIdx/8 owns batch rows [16g,16g+16);
// WG j = blockIdx%8 owns h-slice [32j,32j+32).
// ---------------------------------------------------------------------------
#define LDS_WSK 0u        // 128KB  W_skip^T frags (full)
#define LDS_WHH 131072u   // 64KB   W_hh^T slice frags (8 local ntiles)
#define LDS_ASK 196608u   // 16KB   skip A frags, double buffered (async dst)
#define LDS_AH  212992u   // 8KB    h_in A frags
#define LDS_S   221184u   // 16KB   per-wave skip GEMM result [8][16][32] f32
#define LDS_G   237568u   // 8KB    gates [4][16][32] f32
#define LDS_BSK 245760u   // 1KB    b_skip
#define LDS_TOT 246784u

__global__ __launch_bounds__(256, 1) void rec_kernel(
    const float* __restrict__ b_skip_in, const float* __restrict__ c0,
    unsigned char* __restrict__ ws, float* __restrict__ out) {
  extern __shared__ unsigned char smem[];
  unsigned short* sWsk = (unsigned short*)(smem + LDS_WSK);
  unsigned short* sWhh = (unsigned short*)(smem + LDS_WHH);
  unsigned short* sAsk = (unsigned short*)(smem + LDS_ASK);
  unsigned short* sAh  = (unsigned short*)(smem + LDS_AH);
  float*          sS   = (float*)(smem + LDS_S);
  float*          sG   = (float*)(smem + LDS_G);
  float*          sBsk = (float*)(smem + LDS_BSK);

  const unsigned short* gWsk = (const unsigned short*)(ws + OFF_WSK);
  const unsigned short* gWhh = (const unsigned short*)(ws + OFF_WHH);
  const unsigned short* xgf  = (const unsigned short*)(ws + OFF_XG);
  unsigned short* ring = (unsigned short*)(ws + OFF_RING);
  unsigned int*   ctr  = (unsigned int*)(ws + OFF_CTR);

  int tid  = threadIdx.x;
  int w    = tid >> 5, lane = tid & 31;
  int j    = blockIdx.x & 7;       // h-slice
  int g    = blockIdx.x >> 3;      // batch group
  int b0   = g * 16;

  // --- async-load LDS-resident weights (fragment-swizzled) ---
  for (int i = tid; i < 8192; i += 256)      // full W_skip^T, 128KB
    async_copy_b128((unsigned char*)sWsk + i * 16,
                    (const unsigned char*)gWsk + i * 16);
  for (int i = tid; i < 4096; i += 256) {    // W_hh^T slice, 64KB
    int frag = i >> 6, q = i & 63;
    int l = frag >> 3, kt = frag & 7;
    int NT = (l >> 1) * 16 + 2 * j + (l & 1);
    async_copy_b128((unsigned char*)sWhh + i * 16,
                    (const unsigned char*)gWhh + ((NT * 8 + kt) * 64 + q) * 16);
  }
  for (int i = tid; i < 64; i += 256)        // b_skip, 1KB
    async_copy_b128((unsigned char*)sBsk + i * 16,
                    (const unsigned char*)b_skip_in + i * 16);
  wait_async0();
  __syncthreads();

  // fixed thread->element map: elements (m, lc) and (m+8, lc)
  int lc    = tid & 31;
  int hcol  = 32 * j + lc;
  float creg0 = c0[hcol];
  float creg1 = c0[hcol];

  for (int t = 0; t < Tt; ++t) {
    // --- consume prefetched skip operand ---
    if (t >= WIN) {
      wait_async0();
      __syncthreads();
    }

    // --- skip GEMM: wave w produces S columns [32w, 32w+32) ---
    float* Sw = sS + w * 512;                // [16][32]
    if (t >= WIN) {
      const unsigned short* sAskP = sAsk + (t & 1) * 4096;
      v8f acc0 = {}, acc1 = {};
#pragma unroll
      for (int kt = 0; kt < 8; ++kt) {
        v16bf a   = *(const v16bf*)(sAskP + kt * 512 + lane * 16);
        v16bf bm0 = *(const v16bf*)(sWsk + ((2 * w) * 8 + kt) * 512 + lane * 16);
        v16bf bm1 = *(const v16bf*)(sWsk + ((2 * w + 1) * 8 + kt) * 512 + lane * 16);
        acc0 = wmma_bf16(a, bm0, acc0);
        acc1 = wmma_bf16(a, bm1, acc1);
      }
      int n0 = lane & 15;
#pragma unroll
      for (int r = 0; r < 8; ++r) {
        int m = r + ((lane >> 4) << 3);
        Sw[m * 32 + n0]      = acc0[r];
        Sw[m * 32 + 16 + n0] = acc1[r];
      }
    }

    // --- build h_in A fragment w (k in [32w,32w+32)) from ring entry t-1 ---
    {
      const unsigned short* hsrc =
          ring + ((long)(((t + 31) & 31) * 16 + g)) * 4096 + w * 512 + lane * 16;
      V16U hv;
      hv.q[0] = *(const uint4*)hsrc;
      hv.q[1] = *(const uint4*)(hsrc + 8);
      if (t >= WIN) {
        V16U uv;
        int m = lane & 15;
#pragma unroll
        for (int v2 = 0; v2 < 16; ++v2) {
          int kk = kk_of(lane, v2);
          float val = bf2f(hv.u[v2]) + Sw[m * 32 + kk] + sBsk[32 * w + kk];
          uv.u[v2] = f2bf(val);
        }
        *(v16bf*)(sAh + w * 512 + lane * 16) = uv.v;
      } else {
        *(uint4*)(sAh + w * 512 + lane * 16)     = hv.q[0];
        *(uint4*)(sAh + w * 512 + lane * 16 + 8) = hv.q[1];
      }
    }
    __syncthreads();

    // --- gates GEMM: wave w -> gate ntile NT = (w>>1)*16 + 2j + (w&1) ---
    {
      int chunk = w >> 1, s = w & 1;
      int NT = chunk * 16 + 2 * j + s;
      v8f acc = {};
#pragma unroll
      for (int kt = 0; kt < 8; ++kt) {
        v16bf a  = *(const v16bf*)(sAh + kt * 512 + lane * 16);
        v16bf bm = *(const v16bf*)(sWhh + (w * 8 + kt) * 512 + lane * 16);
        acc = wmma_bf16(a, bm, acc);
      }
      long off = (((long)t * 16 + g) * 64 + NT) * 256 + lane * 8;
      V8U xq;
      xq.q = *(const uint4*)(xgf + off);
      if (t < Tt - 1)
        __builtin_prefetch(xgf + off + (long)16 * 64 * 256, 0, 1);
      int n = lane & 15;
#pragma unroll
      for (int r = 0; r < 8; ++r) {
        int m = r + ((lane >> 4) << 3);
        unsigned wv = (r & 1) ? 0u : 0u;  // placeholder (optimized below)
        float xv = (r & 1) ? bfw_hi(((unsigned*)&xq.q)[r >> 1])
                           : bfw_lo(((unsigned*)&xq.q)[r >> 1]);
        (void)wv;
        sG[(chunk * 16 + m) * 32 + 16 * s + n] = acc[r] + xv;
      }
    }
    __syncthreads();

    // --- LSTM elementwise: 2 elements per thread, c in registers ---
#pragma unroll
    for (int e = 0; e < 2; ++e) {
      int el = tid + e * 256;
      int m  = el >> 5;                      // 0..15
      float gi = sG[(0 * 16 + m) * 32 + lc];
      float gf = sG[(1 * 16 + m) * 32 + lc];
      float gg = sG[(2 * 16 + m) * 32 + lc];
      float go = sG[(3 * 16 + m) * 32 + lc];
      float c_old = e ? creg1 : creg0;
      float cn = fsigmoid(gf) * c_old + fsigmoid(gi) * ftanh(gg);
      float hn = fsigmoid(go) * ftanh(cn);
      if (e) creg1 = cn; else creg0 = cn;
      int b = b0 + m;
      out[((long)b * Tt + t) * Hh + hcol] = hn;
      // store h_new into the bf16 fragment ring (entry t & 31)
      int lane2 = m + (((lc >> 3) & 1) << 4);
      int vv    = (((lc >> 4) & 1) << 3) + (lc & 7);
      ring[((long)((t & 31) * 16 + g)) * 4096 + j * 512 + lane2 * 16 + vv] = f2bf(hn);
      if (t == Tt - 1) {
        out[(long)Bb * Tt * Hh + b * Hh + hcol] = hn;
        out[(long)Bb * Tt * Hh + Bb * Hh + b * Hh + hcol] = cn;
      }
    }

    // --- async-prefetch next step's skip operand (ring entry t+1-24) ---
    if (t >= WIN - 1 && t < Tt - 1) {
      int tn = t + 1;
      const unsigned short* src =
          ring + ((long)(((tn - WIN) & 31) * 16 + g)) * 4096 + tid * 16;
      unsigned short* dst = sAsk + (tn & 1) * 4096 + tid * 16;
      async_copy_b128(dst, src);
      async_copy_b128(dst + 8, src + 8);
    }

    // --- per-step group barrier (8 WGs) ---
    __syncthreads();
    __threadfence();
    if (tid == 0) {
      atomicAdd(&ctr[g], 1u);
      unsigned int target = 8u * (unsigned)(t + 1);
      volatile unsigned int* vc = ctr + g;
      while (*vc < target) __builtin_amdgcn_s_sleep(2);
    }
    __syncthreads();
    __threadfence();
  }
}

// ---------------------------------------------------------------------------
extern "C" void kernel_launch(void* const* d_in, const int* in_sizes, int n_in,
                              void* d_out, int out_size, void* d_ws, size_t ws_size,
                              hipStream_t stream) {
  const float* x      = (const float*)d_in[0];
  const float* W_ih   = (const float*)d_in[1];
  const float* W_hh   = (const float*)d_in[2];
  const float* b_ih   = (const float*)d_in[3];
  const float* b_hh   = (const float*)d_in[4];
  const float* W_skip = (const float*)d_in[5];
  const float* b_skip = (const float*)d_in[6];
  const float* h0     = (const float*)d_in[7];
  const float* c0     = (const float*)d_in[8];
  unsigned char* ws   = (unsigned char*)d_ws;
  float* out          = (float*)d_out;

  // Phase 1: swizzle weights, fold biases, init ring/counters
  swz_init_kernel<<<2053, 256, 0, stream>>>(W_ih, W_hh, b_ih, b_hh, W_skip, h0, ws);

  // Phase 2: xg = x @ W_ih^T + bias (T * 16 * 4 blocks)
  xg_kernel<<<Tt * 16 * 4, 256, 4096, stream>>>(x, ws);

  // Phase 3: persistent recurrent scan (16 groups x 8 WGs, ~241KB LDS each)
  rec_kernel<<<128, 256, LDS_TOT, stream>>>(b_skip, c0, ws, out);
}
